// SplittingMLPNetwork_11570641896173
// MI455X (gfx1250) — matrix operand order repaired and checked by reading, not compiled, not run on hardware
//
#include <hip/hip_runtime.h>

// ---------------------------------------------------------------------------
// SplittingMLPNetwork on MI455X (gfx1250): counting-sort tokens by task so
// every 128-token block uses one weight copy per layer, then run the 3-layer
// MLP with v_wmma_f32_16x16x32_bf16, activations staged in LDS, weights
// pre-transposed to bf16 (L2-resident). Branch-free HW-transcendental tanh.
// ---------------------------------------------------------------------------

#define NTOK     131072
#define DIN      128
#define DHID     256
#define DOUT     64
#define NTASKS   16
#define NCOPIES  4
#define TOKBLK   128   // tokens per workgroup (8 wave32 x 16 rows)

typedef __attribute__((ext_vector_type(16))) __bf16 v16bf;
typedef __attribute__((ext_vector_type(8)))  __bf16 v8bf;
typedef __attribute__((ext_vector_type(8)))  float  v8f;
typedef __attribute__((ext_vector_type(4)))  float  f32x4;

union ABFrag { v8bf h[2]; v16bf v; };

// Branch-free tanh from HW transcendentals:
//   tanh(x) = sign(x) * (1 - e) / (1 + e),  e = exp2(-2*log2(e)*|x|)
// e <= 1 always (no overflow), v_exp_f32 + v_rcp_f32, zero divergence.
__device__ inline float fast_tanh(float x) {
    float ax = __builtin_fabsf(x);
    float u  = ax * -2.8853900817779268f;   // -2 * log2(e)
#if __has_builtin(__builtin_amdgcn_exp2f)
    float e = __builtin_amdgcn_exp2f(u);
#else
    float e = __expf(u * 0.6931471805599453f * 2.0f); // fallback (never taken)
#endif
#if __has_builtin(__builtin_amdgcn_rcpf)
    float r = (1.0f - e) * __builtin_amdgcn_rcpf(1.0f + e);
#else
    float r = (1.0f - e) / (1.0f + e);
#endif
    return __builtin_copysignf(r, x);
}

// Convert 8 contiguous f32 -> v8bf (two b128 global loads + cvt)
__device__ inline v8bf cvt8(const float* __restrict__ p) {
    f32x4 a = *(const f32x4*)p;
    f32x4 b = *(const f32x4*)(p + 4);
    v8bf r;
#pragma unroll
    for (int i = 0; i < 4; ++i) { r[i] = (__bf16)a[i]; r[i + 4] = (__bf16)b[i]; }
    return r;
}

// ---------------------------------------------------------------------------
// Prep kernels: histogram / prefix / scatter (counting sort by task) and
// weight transpose+convert to bf16.
// ---------------------------------------------------------------------------
__global__ void k_zero(int* p, int n) {
    int i = blockIdx.x * blockDim.x + threadIdx.x;
    if (i < n) p[i] = 0;
}

__global__ void k_count(const int* __restrict__ task, int* __restrict__ counts, int n) {
    int i = blockIdx.x * blockDim.x + threadIdx.x;
    if (i < n) atomicAdd(&counts[task[i]], 1);
}

__global__ void k_scan(const int* __restrict__ counts, int* __restrict__ offs,
                       int* __restrict__ cursors) {
    if (threadIdx.x == 0) {
        int acc = 0;
        for (int t = 0; t < NTASKS; ++t) {
            offs[t] = acc;
            cursors[t] = acc;
            acc += counts[t];
        }
    }
}

__global__ void k_scatter(const int* __restrict__ task, int* __restrict__ cursors,
                          int* __restrict__ perm, int n) {
    int i = blockIdx.x * blockDim.x + threadIdx.x;
    if (i < n) {
        int pos = atomicAdd(&cursors[task[i]], 1);
        perm[pos] = i;
    }
}

// src[C][R][Ccols] f32 (row-major, R = K dim) -> dst[C][Ccols][R] bf16
__global__ void k_cvt_wT(const float* __restrict__ src, unsigned short* __restrict__ dst,
                         int C, int R, int Ccols) {
    int i = blockIdx.x * blockDim.x + threadIdx.x;
    int tot = C * R * Ccols;
    if (i >= tot) return;
    int c   = i / (R * Ccols);
    int rem = i % (R * Ccols);
    int r   = rem / Ccols;
    int col = rem % Ccols;
    __bf16 h = (__bf16)src[i];
    dst[(size_t)c * R * Ccols + (size_t)col * R + r] = __builtin_bit_cast(unsigned short, h);
}

// ---------------------------------------------------------------------------
// Fused 3-layer routed MLP. grid = (maxChunks, NTASKS), block = 256.
// Dynamic LDS: H0[128][256] bf16 + H1[128][256] bf16 + tok[128] int.
// ---------------------------------------------------------------------------
__global__ void __launch_bounds__(256)
k_mlp(const float* __restrict__ inputs, const int* __restrict__ copy_maps,
      const float* __restrict__ b0, const float* __restrict__ b1,
      const float* __restrict__ b2,
      const unsigned short* __restrict__ W0t_u, const unsigned short* __restrict__ W1t_u,
      const unsigned short* __restrict__ W2t_u,
      const int* __restrict__ perm, const int* __restrict__ counts,
      const int* __restrict__ offs, float* __restrict__ out) {
    const int t      = blockIdx.y;
    const int segCnt = counts[t];
    const int within = blockIdx.x * TOKBLK;
    if (within >= segCnt) return;
    const int base = offs[t] + within;
    const int cnt  = min(TOKBLK, segCnt - within);

    const int c0 = copy_maps[t];
    const int c1 = copy_maps[NTASKS + t];
    const int c2 = copy_maps[2 * NTASKS + t];

    const __bf16* W0t = (const __bf16*)W0t_u;
    const __bf16* W1t = (const __bf16*)W1t_u;
    const __bf16* W2t = (const __bf16*)W2t_u;

    extern __shared__ char smem_raw[];
    __bf16* H0 = (__bf16*)smem_raw;           // 128*256 bf16 = 64 KB
    __bf16* H1 = H0 + TOKBLK * DHID;          // 128*256 bf16 = 64 KB
    int*    tok = (int*)(H1 + TOKBLK * DHID); // 128 ints

    for (int r = threadIdx.x; r < TOKBLK; r += blockDim.x)
        tok[r] = perm[base + min(r, cnt - 1)];
    __syncthreads();

    const int lane    = threadIdx.x & 31;
    const int wv      = threadIdx.x >> 5;     // wave id: row tile 0..7
    const int rowBase = wv * 16;
    const int mrow    = rowBase + (lane & 15);
    const int koff    = (lane < 16) ? 0 : 8;  // A-fragment K offset per ISA layout
    const int kboff   = (lane < 16) ? 0 : 16; // unused; B fragments load with koff
    const int ncol    = lane & 15;            // C/D column = lane % 16

    const float* xrow = inputs + (size_t)tok[mrow] * DIN;

    // ---- Layer 0: [cnt x 128] @ W0[c0] (128x256) + b0, tanh -> H0 ----
    {
        const __bf16* w = W0t + (size_t)c0 * DHID * DIN;
        for (int j = 0; j < DHID / 16; ++j) {
            const __bf16* wcol = w + (size_t)(j * 16 + ncol) * DIN;
            __builtin_prefetch(wcol, 0, 1);
            v8f acc = {};
#pragma unroll
            for (int kk = 0; kk < DIN; kk += 32) {
                ABFrag a, b;
                a.h[0] = cvt8(xrow + kk + koff);
                a.h[1] = cvt8(xrow + kk + 16 + koff);
                b.h[0] = *(const v8bf*)(wcol + kk + koff);
                b.h[1] = *(const v8bf*)(wcol + kk + 16 + koff);
                acc = __builtin_amdgcn_wmma_f32_16x16x32_bf16(
                    false, a.v, false, b.v, (short)0, acc, false, false);
            }
            const float bias = b0[c0 * DHID + j * 16 + ncol];
#pragma unroll
            for (int v = 0; v < 8; ++v) {
                const int m = rowBase + ((lane < 16) ? v : v + 8);
                H0[m * DHID + j * 16 + ncol] = (__bf16)fast_tanh(acc[v] + bias);
            }
        }
    }

    // ---- Layer 1: H0 @ W1[c1] (256x256) + b1, tanh -> H1 ----
    {
        const __bf16* w    = W1t + (size_t)c1 * DHID * DHID;
        const __bf16* hrow = H0 + (size_t)mrow * DHID;
        for (int j = 0; j < DHID / 16; ++j) {
            const __bf16* wcol = w + (size_t)(j * 16 + ncol) * DHID;
            __builtin_prefetch(wcol, 0, 1);
            v8f acc = {};
#pragma unroll
            for (int kk = 0; kk < DHID; kk += 32) {
                ABFrag a, b;
                a.h[0] = *(const v8bf*)(hrow + kk + koff);
                a.h[1] = *(const v8bf*)(hrow + kk + 16 + koff);
                b.h[0] = *(const v8bf*)(wcol + kk + koff);
                b.h[1] = *(const v8bf*)(wcol + kk + 16 + koff);
                acc = __builtin_amdgcn_wmma_f32_16x16x32_bf16(
                    false, a.v, false, b.v, (short)0, acc, false, false);
            }
            const float bias = b1[c1 * DHID + j * 16 + ncol];
#pragma unroll
            for (int v = 0; v < 8; ++v) {
                const int m = rowBase + ((lane < 16) ? v : v + 8);
                H1[m * DHID + j * 16 + ncol] = (__bf16)fast_tanh(acc[v] + bias);
            }
        }
    }

    // ---- Layer 2: H1 @ W2[c2] (256x64) + b2 -> out (gathered scatter) ----
    {
        const __bf16* w    = W2t + (size_t)c2 * DOUT * DHID;
        const __bf16* hrow = H1 + (size_t)mrow * DHID;
        for (int j = 0; j < DOUT / 16; ++j) {
            const __bf16* wcol = w + (size_t)(j * 16 + ncol) * DHID;
            v8f acc = {};
#pragma unroll
            for (int kk = 0; kk < DHID; kk += 32) {
                ABFrag a, b;
                a.h[0] = *(const v8bf*)(hrow + kk + koff);
                a.h[1] = *(const v8bf*)(hrow + kk + 16 + koff);
                b.h[0] = *(const v8bf*)(wcol + kk + koff);
                b.h[1] = *(const v8bf*)(wcol + kk + 16 + koff);
                acc = __builtin_amdgcn_wmma_f32_16x16x32_bf16(
                    false, a.v, false, b.v, (short)0, acc, false, false);
            }
            const float bias = b2[c2 * DOUT + j * 16 + ncol];
#pragma unroll
            for (int v = 0; v < 8; ++v) {
                const int m = rowBase + ((lane < 16) ? v : v + 8);
                if (m < cnt)
                    out[(size_t)tok[m] * DOUT + j * 16 + ncol] = acc[v] + bias;
            }
        }
    }
}

// ---------------------------------------------------------------------------
// Host launcher
// ---------------------------------------------------------------------------
extern "C" void kernel_launch(void* const* d_in, const int* in_sizes, int n_in,
                              void* d_out, int out_size, void* d_ws, size_t ws_size,
                              hipStream_t stream) {
    const float* inputs = (const float*)d_in[0];
    const int*   task   = (const int*)d_in[1];
    const int*   cmap   = (const int*)d_in[2];
    const float* W0     = (const float*)d_in[3];
    const float* b0     = (const float*)d_in[4];
    const float* W1     = (const float*)d_in[5];
    const float* b1     = (const float*)d_in[6];
    const float* W2     = (const float*)d_in[7];
    const float* b2     = (const float*)d_in[8];
    float*       out    = (float*)d_out;

    const int N = in_sizes[1]; // token count (task_indices length)

    // Workspace layout (all offsets 16B-aligned)
    char* ws = (char*)d_ws;
    int* counts  = (int*)(ws);          // 16 ints
    int* offs    = (int*)(ws + 64);     // 16 ints
    int* cursors = (int*)(ws + 128);    // 16 ints
    int* perm    = (int*)(ws + 256);    // N ints
    unsigned short* W0t = (unsigned short*)(ws + 256 + (size_t)N * 4);
    unsigned short* W1t = W0t + (size_t)NCOPIES * DHID * DIN;   // 4*256*128
    unsigned short* W2t = W1t + (size_t)NCOPIES * DHID * DHID;  // 4*256*256

    // 1) counting sort by task
    k_zero<<<1, 64, 0, stream>>>((int*)ws, 48);
    k_count<<<(N + 255) / 256, 256, 0, stream>>>(task, counts, N);
    k_scan<<<1, 32, 0, stream>>>(counts, offs, cursors);
    k_scatter<<<(N + 255) / 256, 256, 0, stream>>>(task, cursors, perm, N);

    // 2) weights -> transposed bf16 (L2-resident, ~0.9 MB total)
    {
        int tot0 = NCOPIES * DIN * DHID;
        int tot1 = NCOPIES * DHID * DHID;
        int tot2 = NCOPIES * DHID * DOUT;
        k_cvt_wT<<<(tot0 + 255) / 256, 256, 0, stream>>>(W0, W0t, NCOPIES, DIN, DHID);
        k_cvt_wT<<<(tot1 + 255) / 256, 256, 0, stream>>>(W1, W1t, NCOPIES, DHID, DHID);
        k_cvt_wT<<<(tot2 + 255) / 256, 256, 0, stream>>>(W2, W2t, NCOPIES, DHID, DOUT);
    }

    // 3) fused 3-layer WMMA MLP over sorted segments
    const int maxChunks = (N + TOKBLK - 1) / TOKBLK;
    dim3 grid(maxChunks, NTASKS);
    size_t smem = (size_t)2 * TOKBLK * DHID * sizeof(unsigned short) // H0 + H1
                + (size_t)TOKBLK * sizeof(int);                      // tok[]
    k_mlp<<<grid, 256, smem, stream>>>(inputs, cmap, b0, b1, b2,
                                       W0t, W1t, W2t,
                                       perm, counts, offs, out);
}
